// Model_3487513444700
// MI455X (gfx1250) — matrix operand order repaired and checked by reading, not compiled, not run on hardware
//
#include <hip/hip_runtime.h>

#define NBK 4
#define BB  8
#define TT  1024
#define DD  1024
#define VV  256
#define II  2816
#define EPSF 1e-6f

typedef __bf16 v16bf __attribute__((ext_vector_type(16)));
typedef __bf16 v8bf  __attribute__((ext_vector_type(8)));
typedef float  v8f   __attribute__((ext_vector_type(8)));
typedef unsigned int u32x4 __attribute__((ext_vector_type(4)));
typedef int  i32x4 __attribute__((ext_vector_type(4)));
typedef int  i32x8 __attribute__((ext_vector_type(8)));

// ---------------- reductions (wave32) ----------------
__device__ __forceinline__ float sigf(float x){ return 1.f/(1.f+expf(-x)); }

__device__ __forceinline__ float warp_sum(float v){
  #pragma unroll
  for(int o=16;o>0;o>>=1) v += __shfl_down(v,o,32);
  return v;
}
__device__ __forceinline__ float block_sum(float v, float* s){
  int lane=threadIdx.x&31, w=threadIdx.x>>5;
  v=warp_sum(v);
  __syncthreads();
  if(lane==0) s[w]=v;
  __syncthreads();
  float r=(threadIdx.x<8)?s[threadIdx.x]:0.f;
  r=warp_sum(r);
  if(threadIdx.x==0) s[0]=r;
  __syncthreads();
  return s[0];
}
__device__ __forceinline__ float warp_max(float v){
  #pragma unroll
  for(int o=16;o>0;o>>=1) v=fmaxf(v,__shfl_down(v,o,32));
  return v;
}
__device__ __forceinline__ float block_max(float v, float* s){
  int lane=threadIdx.x&31, w=threadIdx.x>>5;
  v=warp_max(v);
  __syncthreads();
  if(lane==0) s[w]=v;
  __syncthreads();
  float r=(threadIdx.x<8)?s[threadIdx.x]:-3.4e38f;
  r=warp_max(r);
  if(threadIdx.x==0) s[0]=r;
  __syncthreads();
  return s[0];
}

// ---------------- TDM staging: global 2D bf16 tile [rows x cols] -> LDS ----------------
// D# per CDNA5 ISA: group0 = {count=1, lds_addr, global_addr(57b), type=2};
// group1 = {data_size=2B, tensor_dim0=cols, tensor_dim1=rows, tile_dim0=cols, tile_dim1=rows,
//           tensor_dim0_stride=cols}. Groups 2/3 zero (2D tensor).
__device__ __forceinline__ void stage_A_lds(const __bf16* __restrict__ gA, __bf16* shA,
                                            unsigned rows, unsigned cols){
#if __has_builtin(__builtin_amdgcn_tensor_load_to_lds) && __has_builtin(__builtin_amdgcn_s_wait_tensorcnt)
  if(threadIdx.x == 0){
    unsigned long long ga = (unsigned long long)(size_t)(const void*)gA;
    unsigned lds_addr = (unsigned)(size_t)(void*)shA;   // low 32 bits of generic = LDS offset
    u32x4 g0 = { 1u,                                    // count=1, user mode, no gather
                 lds_addr,
                 (unsigned)(ga & 0xffffffffu),
                 (unsigned)((ga >> 32) & 0x01ffffffu) | (2u << 30) };  // addr[56:32] | type=2
    i32x8 g1;
    g1[0] = (int)(1u << 16);                                  // wg_mask=0, data_size=1 (2 bytes)
    g1[1] = (int)((cols & 0xffffu) << 16);                    // tensor_dim0[15:0]
    g1[2] = (int)(((cols >> 16) & 0xffffu) | ((rows & 0xffffu) << 16)); // dim0[31:16] | dim1[15:0]
    g1[3] = (int)(((rows >> 16) & 0xffffu) | ((cols & 0xffffu) << 16)); // dim1[31:16] | tile_dim0
    g1[4] = (int)(rows & 0xffffu);                            // tile_dim1 | tile_dim2=0
    g1[5] = (int)cols;                                        // tensor_dim0_stride[31:0]
    g1[6] = 0;                                                // stride0 hi | stride1 lo
    g1[7] = 0;                                                // stride1 hi
    i32x4 z4 = {0,0,0,0};
#if defined(__clang_major__) && (__clang_major__ >= 23)
    i32x8 z8 = {0,0,0,0,0,0,0,0};
    __builtin_amdgcn_tensor_load_to_lds(g0, g1, z4, z4, z8, 0);
#else
    __builtin_amdgcn_tensor_load_to_lds(g0, g1, z4, z4, 0);
#endif
    __builtin_amdgcn_s_wait_tensorcnt((short)0);   // issuing wave waits for TDM completion
  }
#else
  // fallback: cooperative 16B-chunk copy
  const v8bf* src = (const v8bf*)gA;
  v8bf* dst = (v8bf*)shA;
  int n8 = (int)((rows*cols) >> 3);
  for(int i=(int)threadIdx.x; i<n8; i+=(int)blockDim.x) dst[i] = src[i];
#endif
  __syncthreads();
}

// ---------------- WMMA core: D[16x16] = A[16xK] * W[N x K]^T tile ----------------
// A row-major bf16 [16,K] (LDS or global); W row-major bf16 [N,K] -> B cols are W rows (contiguous K).
// A frag: lane<16: M=lane, K octets {0..7,16..23}; lane>=16: M=lane-16, {8..15,24..31}.
// B frag: lane%16 = column N, same K octets as A: lanes<16 {0..7,16..23}, lanes>=16 {8..15,24..31}.
__device__ __forceinline__ v8f wmma_rowmajor_tile(const __bf16* __restrict__ A,
                                                  const __bf16* __restrict__ W,
                                                  int K, int n0){
  const int lane = threadIdx.x & 31;
  const int sub  = lane & 15;
  const int hi   = lane >> 4;
  const __bf16* arow = A + (size_t)sub * K + hi*8;
  const __bf16* brow = W + (size_t)(n0 + sub) * K + hi*8;
  v8f acc = {};
  #pragma unroll 4
  for(int k0=0;k0<K;k0+=32){
    __builtin_prefetch((const void*)(brow + k0 + 512), 0, 3); // near-cache prefetch of weights
    v8bf a0 = *(const v8bf*)(arow + k0);
    v8bf a1 = *(const v8bf*)(arow + k0 + 16);
    v8bf b0 = *(const v8bf*)(brow + k0);
    v8bf b1 = *(const v8bf*)(brow + k0 + 16);
    v16bf av = __builtin_shufflevector(a0,a1,0,1,2,3,4,5,6,7,8,9,10,11,12,13,14,15);
    v16bf bv = __builtin_shufflevector(b0,b1,0,1,2,3,4,5,6,7,8,9,10,11,12,13,14,15);
    acc = __builtin_amdgcn_wmma_f32_16x16x32_bf16(false, av, false, bv,
                                                  (short)0, acc, false, false);
  }
  return acc;
}

// ---------------- one-time kernels ----------------
__global__ void cvt_bf16_kernel(const float* __restrict__ s, __bf16* __restrict__ d, long n){
  long i = (long)blockIdx.x*blockDim.x + threadIdx.x;
  long st = (long)gridDim.x*blockDim.x;
  for(; i<n; i+=st) d[i] = (__bf16)s[i];
}
__global__ void zero_f32_kernel(float* p, long n){
  long i = (long)blockIdx.x*blockDim.x + threadIdx.x;
  long st = (long)gridDim.x*blockDim.x;
  for(; i<n; i+=st) p[i] = 0.f;
}
// lb[i] = cumsum(softmax(weight_l[i])) - weight_l[i,0]
__global__ void lb_kernel(const float* __restrict__ wl, float* __restrict__ lb){
  __shared__ float s[8];
  __shared__ float e[DD];
  int nb = blockIdx.x, tid = threadIdx.x;
  const float* w = wl + (size_t)nb*DD;
  float vv[4], mx=-3.4e38f;
  #pragma unroll
  for(int j=0;j<4;j++){ vv[j]=w[tid+256*j]; mx=fmaxf(mx,vv[j]); }
  mx = block_max(mx, s);
  float ss=0.f;
  #pragma unroll
  for(int j=0;j<4;j++){ float ev=expf(vv[j]-mx); e[tid+256*j]=ev; ss+=ev; }
  ss = block_sum(ss, s);
  __syncthreads();
  if(tid==0){ float c=0.f; for(int d=0; d<DD; d++){ c+=e[d]; e[d]=c; } }
  __syncthreads();
  float w0 = w[0];
  #pragma unroll
  for(int j=0;j<4;j++){ int d=tid+256*j; lb[(size_t)nb*DD+d] = e[d]/ss - w0; }
}

// ---------------- per-step per-row kernels (blockIdx.x = batch row 0..7) ----------------
__global__ void pre3_kernel(float* __restrict__ ybuf, const float* __restrict__ emb,
                            const int* __restrict__ x, int t, int gather,
                            const float* __restrict__ wgin, const float* __restrict__ wgf,
                            const float* __restrict__ wgc, const float* __restrict__ wgg,
                            __bf16* __restrict__ A3){
  __shared__ float s[8];
  int b = blockIdx.x, tid = threadIdx.x;
  float yv[4], xn[4];
  float ss=0.f;
  if(gather){
    int tok = x[(size_t)b*TT + t];
    #pragma unroll
    for(int j=0;j<4;j++){ int d=tid+256*j; float v=emb[(size_t)tok*DD+d];
                          yv[j]=v; ybuf[(size_t)b*DD+d]=v; ss+=v*v; }
  } else {
    #pragma unroll
    for(int j=0;j<4;j++){ int d=tid+256*j; float v=ybuf[(size_t)b*DD+d]; yv[j]=v; ss+=v*v; }
  }
  float r1 = rsqrtf(block_sum(ss,s)*(1.f/DD) + EPSF);
  float ss2=0.f;
  #pragma unroll
  for(int j=0;j<4;j++){ int d=tid+256*j; float v=yv[j]*r1*wgin[d]; xn[j]=v; ss2+=v*v; }
  float r2 = rsqrtf(block_sum(ss2,s)*(1.f/DD) + EPSF);
  #pragma unroll
  for(int j=0;j<4;j++){
    int d=tid+256*j; float v=xn[j]*r2;
    A3[(size_t)b*DD+d]      = (__bf16)(v*wgf[d]);
    A3[(size_t)(16+b)*DD+d] = (__bf16)(v*wgc[d]);
    A3[(size_t)(32+b)*DD+d] = (__bf16)(v*wgg[d]);
  }
}

// h update + output-gate RMS; FCG already contains bias from the GEMM epilogue.
__global__ void h_update_kernel(const float* __restrict__ FCG,
                                const float* __restrict__ lbv,
                                const float* __restrict__ wnorm, const float* __restrict__ wout,
                                float* __restrict__ hst, __bf16* __restrict__ Ao){
  __shared__ float s[8];
  int b = blockIdx.x, tid = threadIdx.x;
  float fr[4], cr[4], gr[4], vv[4];
  float ssg=0.f;
  #pragma unroll
  for(int j=0;j<4;j++){
    int d=tid+256*j;
    fr[j]=FCG[(size_t)b*DD+d];
    cr[j]=FCG[(size_t)(16+b)*DD+d];
    float g=FCG[(size_t)(32+b)*DD+d];
    gr[j]=g; ssg+=g*g;
  }
  float rg = rsqrtf(block_sum(ssg,s)*(1.f/DD) + EPSF);
  float ssv=0.f;
  #pragma unroll
  for(int j=0;j<4;j++){
    int d=tid+256*j;
    float xg = gr[j]*rg*wnorm[d];
    float xf = sigf(fr[j]);
    float xc = cr[j]*sigf(cr[j]);          // silu
    float l  = lbv[d];
    float f  = l + (1.f-l)*xf;
    float hn = f*hst[(size_t)b*DD+d] + (1.f-f)*xc;
    hst[(size_t)b*DD+d] = hn;
    float v  = xg*hn*sigf(hn);
    vv[j]=v; ssv+=v*v;
  }
  float rv = rsqrtf(block_sum(ssv,s)*(1.f/DD) + EPSF);
  #pragma unroll
  for(int j=0;j<4;j++){ int d=tid+256*j; Ao[(size_t)b*DD+d]=(__bf16)(vv[j]*rv*wout[d]); }
}

__global__ void glu_pre_kernel(const float* __restrict__ Xo,
                               const float* __restrict__ gwg, const float* __restrict__ gwgo,
                               __bf16* __restrict__ Au){
  __shared__ float s[8];
  int b = blockIdx.x, tid = threadIdx.x;
  float xv[4], tv[4];
  float ss=0.f;
  #pragma unroll
  for(int j=0;j<4;j++){ int d=tid+256*j; float v=Xo[(size_t)b*DD+d]; xv[j]=v; ss+=v*v; }
  float r1 = rsqrtf(block_sum(ss,s)*(1.f/DD) + EPSF);
  float ss2=0.f;
  #pragma unroll
  for(int j=0;j<4;j++){ int d=tid+256*j; float v=xv[j]*r1*gwg[d]; tv[j]=v; ss2+=v*v; }
  float r2 = rsqrtf(block_sum(ss2,s)*(1.f/DD) + EPSF);
  #pragma unroll
  for(int j=0;j<4;j++){ int d=tid+256*j; Au[(size_t)b*DD+d]=(__bf16)(tv[j]*r2*gwgo[d]); }
}

__global__ void glu_mid_kernel(const float* __restrict__ UU,
                               const float* __restrict__ gwd, __bf16* __restrict__ Ad){
  __shared__ float s[8];
  int b = blockIdx.x, tid = threadIdx.x;
  float vv[11];
  float ss=0.f;
  #pragma unroll
  for(int j=0;j<11;j++){
    int d=tid+256*j;
    float u1=UU[(size_t)b*(2*II)+d];
    float u2=UU[(size_t)b*(2*II)+II+d];
    float v = u1*sigf(u1)*u2;
    vv[j]=v; ss+=v*v;
  }
  float r = rsqrtf(block_sum(ss,s)*(1.f/II) + EPSF);
  #pragma unroll
  for(int j=0;j<11;j++){ int d=tid+256*j; Ad[(size_t)b*II+d]=(__bf16)(vv[j]*r*gwd[d]); }
}

// ---------------- GEMM kernels (256 thr = 8 waves, one 16x16 tile per wave) ----------------
// A (16x1024 bf16, 32KB) staged into LDS via TDM; B streamed from L2.
__global__ void gemm3_kernel(const __bf16* __restrict__ A3,
                             const __bf16* __restrict__ Wf, const __bf16* __restrict__ Wc,
                             const __bf16* __restrict__ Wg,
                             const float* __restrict__ bf_, const float* __restrict__ bc_,
                             const float* __restrict__ bg_, float* __restrict__ FCG){
  __shared__ __bf16 shA[16*DD];
  int mm = blockIdx.x >> 3;                              // matmul 0..2 (blocks 0-7,8-15,16-23)
  stage_A_lds(A3 + (size_t)mm*16*DD, shA, 16, DD);
  int tw = (blockIdx.x & 7)*8 + (threadIdx.x >> 5);      // tile 0..63 within this matmul
  int n0 = tw << 4;
  const __bf16* W = (mm==0)?Wf:((mm==1)?Wc:Wg);
  const float*  B = (mm==0)?bf_:((mm==1)?bc_:bg_);
  v8f acc = wmma_rowmajor_tile(shA, W, DD, n0);
  int lane=threadIdx.x&31, sub=lane&15, hi=lane>>4;
  int n = n0 + sub; float bv = B[n];
  float* out = FCG + (size_t)mm*16*DD;
  #pragma unroll
  for(int r=0;r<8;r++){ int m=r+8*hi; out[(size_t)m*DD+n] = acc[r]+bv; }
}

__global__ void gemm_o_kernel(const __bf16* __restrict__ Ao, const __bf16* __restrict__ W,
                              const float* __restrict__ bias, const float* __restrict__ ybuf,
                              float* __restrict__ Xo){
  __shared__ __bf16 shA[16*DD];
  stage_A_lds(Ao, shA, 16, DD);
  int gw = blockIdx.x*8 + (threadIdx.x >> 5);            // 0..63
  int n0 = gw << 4;
  v8f acc = wmma_rowmajor_tile(shA, W, DD, n0);
  int lane=threadIdx.x&31, sub=lane&15, hi=lane>>4;
  int n = n0 + sub; float bv = bias[n];
  #pragma unroll
  for(int r=0;r<8;r++){ int m=r+8*hi; Xo[(size_t)m*DD+n] = acc[r]+bv+ybuf[(size_t)m*DD+n]; }
}

__global__ void gemm_u_kernel(const __bf16* __restrict__ Au, const __bf16* __restrict__ W,
                              const float* __restrict__ bias, float* __restrict__ UU){
  __shared__ __bf16 shA[16*DD];
  stage_A_lds(Au, shA, 16, DD);
  int gw = blockIdx.x*8 + (threadIdx.x >> 5);            // 0..351
  int n0 = gw << 4;
  v8f acc = wmma_rowmajor_tile(shA, W, DD, n0);
  int lane=threadIdx.x&31, sub=lane&15, hi=lane>>4;
  int n = n0 + sub; float bv = bias[n];
  #pragma unroll
  for(int r=0;r<8;r++){ int m=r+8*hi; UU[(size_t)m*(2*II)+n] = acc[r]+bv; }
}

// A = 16x2816 (88KB) -> read A directly from global (stays within static-LDS budget).
__global__ void gemm_n_kernel(const __bf16* __restrict__ Ad, const __bf16* __restrict__ W,
                              const float* __restrict__ bias, const float* __restrict__ Xo,
                              float* __restrict__ ybuf, __bf16* __restrict__ ybf){
  int gw = (blockIdx.x*blockDim.x + threadIdx.x) >> 5;   // 0..63
  int n0 = gw << 4;
  v8f acc = wmma_rowmajor_tile(Ad, W, II, n0);
  int lane=threadIdx.x&31, sub=lane&15, hi=lane>>4;
  int n = n0 + sub; float bv = bias[n];
  #pragma unroll
  for(int r=0;r<8;r++){
    int m=r+8*hi;
    float y = acc[r]+bv+Xo[(size_t)m*DD+n];
    ybuf[(size_t)m*DD+n]=y;
    ybf [(size_t)m*DD+n]=(__bf16)y;
  }
}

__global__ void logits_kernel(const __bf16* __restrict__ ybf, const __bf16* __restrict__ W,
                              const float* __restrict__ bias, float* __restrict__ out, int t){
  __shared__ __bf16 shA[16*DD];
  stage_A_lds(ybf, shA, 16, DD);
  int gw = blockIdx.x*8 + (threadIdx.x >> 5);            // 0..15
  int n0 = gw << 4;
  v8f acc = wmma_rowmajor_tile(shA, W, DD, n0);
  int lane=threadIdx.x&31, sub=lane&15, hi=lane>>4;
  int n = n0 + sub; float bv = bias[n];
  if(hi==0){   // rows 0..7 = real batch rows
    #pragma unroll
    for(int r=0;r<8;r++){
      out[(size_t)r*TT*VV + (size_t)t*VV + n] = acc[r]+bv;
    }
  }
}

// ---------------- host ----------------
extern "C" void kernel_launch(void* const* d_in, const int* in_sizes, int n_in,
                              void* d_out, int out_size, void* d_ws, size_t ws_size,
                              hipStream_t stream) {
  const int*   x      = (const int*)  d_in[0];
  const float* emb    = (const float*)d_in[1];
  const float* wg_in  = (const float*)d_in[2];
  const float* wg_f   = (const float*)d_in[3];
  const float* wg_c   = (const float*)d_in[4];
  const float* wg_g   = (const float*)d_in[5];
  const float* wg_nrm = (const float*)d_in[6];
  const float* lf_w   = (const float*)d_in[7];
  const float* lf_b   = (const float*)d_in[8];
  const float* lc_w   = (const float*)d_in[9];
  const float* lc_b   = (const float*)d_in[10];
  const float* lg_w   = (const float*)d_in[11];
  const float* lg_b   = (const float*)d_in[12];
  const float* w_l    = (const float*)d_in[13];
  const float* wg_oo  = (const float*)d_in[14];
  const float* lo_w   = (const float*)d_in[15];
  const float* lo_b   = (const float*)d_in[16];
  const float* glu_wg = (const float*)d_in[17];
  const float* glu_wgo= (const float*)d_in[18];
  const float* glu_wd = (const float*)d_in[19];
  const float* lu_w   = (const float*)d_in[20];
  const float* lu_b   = (const float*)d_in[21];
  const float* ln_w   = (const float*)d_in[22];
  const float* ln_b   = (const float*)d_in[23];
  const float* out_w  = (const float*)d_in[24];
  const float* out_b  = (const float*)d_in[25];
  float* out = (float*)d_out;

  // ---- workspace bump allocator (256B aligned) ----
  char* p = (char*)d_ws;
  auto alloc = [&](size_t bytes)->char* {
    char* r = p; p += (bytes + 255) & ~(size_t)255; return r;
  };
  const size_t nDxD = (size_t)NBK*DD*DD;
  const size_t nLU  = (size_t)NBK*2*II*DD;
  const size_t nLN  = (size_t)NBK*DD*II;
  const size_t nOW  = (size_t)VV*DD;
  __bf16* lfw = (__bf16*)alloc(nDxD*2);
  __bf16* lcw = (__bf16*)alloc(nDxD*2);
  __bf16* lgw = (__bf16*)alloc(nDxD*2);
  __bf16* low = (__bf16*)alloc(nDxD*2);
  __bf16* luw = (__bf16*)alloc(nLU*2);
  __bf16* lnw = (__bf16*)alloc(nLN*2);
  __bf16* oww = (__bf16*)alloc(nOW*2);
  float*  lbv = (float*) alloc((size_t)NBK*DD*4);
  float*  hst = (float*) alloc((size_t)NBK*16*DD*4);
  float*  ybuf= (float*) alloc((size_t)16*DD*4);
  __bf16* ybf = (__bf16*)alloc((size_t)16*DD*2);
  __bf16* A3  = (__bf16*)alloc((size_t)3*16*DD*2);
  float*  FCG = (float*) alloc((size_t)3*16*DD*4);
  __bf16* Ao  = (__bf16*)alloc((size_t)16*DD*2);
  float*  Xo  = (float*) alloc((size_t)16*DD*4);
  __bf16* Au  = (__bf16*)alloc((size_t)16*DD*2);
  float*  UU  = (float*) alloc((size_t)16*2*II*4);
  __bf16* Ad  = (__bf16*)alloc((size_t)16*II*2);
  (void)ws_size; (void)n_in; (void)in_sizes; (void)out_size;

  // ---- one-time: weights -> bf16 (L2-resident ~103MB), lower-bound vector, h=0 ----
  cvt_bf16_kernel<<<2048,256,0,stream>>>(lf_w, lfw, (long)nDxD);
  cvt_bf16_kernel<<<2048,256,0,stream>>>(lc_w, lcw, (long)nDxD);
  cvt_bf16_kernel<<<2048,256,0,stream>>>(lg_w, lgw, (long)nDxD);
  cvt_bf16_kernel<<<2048,256,0,stream>>>(lo_w, low, (long)nDxD);
  cvt_bf16_kernel<<<2048,256,0,stream>>>(lu_w, luw, (long)nLU);
  cvt_bf16_kernel<<<2048,256,0,stream>>>(ln_w, lnw, (long)nLN);
  cvt_bf16_kernel<<<512, 256,0,stream>>>(out_w, oww, (long)nOW);
  lb_kernel<<<NBK,256,0,stream>>>(w_l, lbv);
  zero_f32_kernel<<<256,256,0,stream>>>(hst, (long)NBK*16*DD);

  // ---- sequential scan over T; stream order enforces stage dependencies ----
  for(int t=0;t<TT;t++){
    for(int i=0;i<NBK;i++){
      size_t oD  = (size_t)i*DD;
      size_t oDD = (size_t)i*DD*DD;
      size_t o2I = (size_t)i*2*II;
      pre3_kernel<<<BB,256,0,stream>>>(ybuf, emb, x, t, (i==0)?1:0,
                                       wg_in+oD, wg_f+oD, wg_c+oD, wg_g+oD, A3);
      gemm3_kernel<<<24,256,0,stream>>>(A3, lfw+oDD, lcw+oDD, lgw+oDD,
                                        lf_b+oD, lc_b+oD, lg_b+oD, FCG);
      h_update_kernel<<<BB,256,0,stream>>>(FCG, lbv+oD,
                                           wg_nrm+oD, wg_oo+oD, hst+(size_t)i*16*DD, Ao);
      gemm_o_kernel<<<8,256,0,stream>>>(Ao, low+oDD, lo_b+oD, ybuf, Xo);
      glu_pre_kernel<<<BB,256,0,stream>>>(Xo, glu_wg+oD, glu_wgo+oD, Au);
      gemm_u_kernel<<<44,256,0,stream>>>(Au, luw+(size_t)i*2*II*DD, lu_b+o2I, UU);
      glu_mid_kernel<<<BB,256,0,stream>>>(UU, glu_wd+(size_t)i*II, Ad);
      gemm_n_kernel<<<8,256,0,stream>>>(Ad, lnw+(size_t)i*DD*II, ln_b+oD, Xo, ybuf, ybf);
    }
    logits_kernel<<<2,256,0,stream>>>(ybf, oww, out_b, out, t);
  }
}
